// V2XGraphPlusPredictor_54958401520078
// MI455X (gfx1250) — matrix operand order repaired and checked by reading, not compiled, not run on hardware
//
#include <hip/hip_runtime.h>
#include <cstddef>

// ---------------------------------------------------------------------------
// V2XGraphPlusPredictor for MI455X (gfx1250, wave32, WMMA f16 16x16x32)
// ---------------------------------------------------------------------------
#define Bsz   128
#define Tlen  50
#define D_IN  4
#define NB    32
#define LL    30
#define PP    20
#define HID   128
#define GRU_H 64
#define NNODE 64      // 1 ego + 32 nbr + 1 infra + 30 lanes
#define FUT   30
#define MODES 6
#define OUTD  2
#define LNEPS 1e-5f

typedef __attribute__((ext_vector_type(16))) _Float16 v16h;
typedef __attribute__((ext_vector_type(4)))  _Float16 v4h;
typedef __attribute__((ext_vector_type(8)))  float    v8f;

__device__ __forceinline__ float geluf(float x) {
  return 0.5f * x * (1.f + erff(x * 0.70710678118654752f));
}
__device__ __forceinline__ float sigmoidf(float x) {
  return 1.f / (1.f + expf(-x));
}

// ---------------------------------------------------------------------------
// Generic WMMA GEMM:  C[M,N] = epi(A[M,K] @ W[N,K]^T + bias) (+ pos[row%posT])
// Block = 128 threads (4 waves). Block tile 64x64, wave tile 16x64, K-step 32.
// Full tiles use float4 (global_load_b128) -> f16x4 (ds_store_b64) staging with
// global_prefetch of the next K-slab; partial tiles fall back to guarded loads.
// epi: 0 none, 1 gelu, 2 relu, 3 sigmoid
// ---------------------------------------------------------------------------
__launch_bounds__(128)
__global__ void gemm_wmma_kernel(const float* __restrict__ A, int lda,
                                 const float* __restrict__ W,
                                 const float* __restrict__ bias,
                                 float* __restrict__ C, int ldc,
                                 int M, int N, int K,
                                 int epi, const float* __restrict__ pos, int posT) {
  // pad 32 -> 36 halves (72B rows): 8B-aligned f16x4 stores, 4B-aligned pair reads
  __shared__ __align__(16) _Float16 As[64][36];
  __shared__ __align__(16) _Float16 Bs[64][36];
  const int tid  = threadIdx.x;
  const int wave = tid >> 5;
  const int lane = tid & 31;
  const int m0 = blockIdx.y * 64;
  const int n0 = blockIdx.x * 64;
  const bool fullA = (m0 + 64 <= M);
  const bool fullB = (n0 + 64 <= N);

  v8f acc[4];
  for (int i = 0; i < 4; ++i)
    for (int r = 0; r < 8; ++r) acc[i][r] = 0.f;

  for (int kb = 0; kb < K; kb += 32) {
    if (fullA) {
      #pragma unroll
      for (int i = 0; i < 4; ++i) {               // 512 float4s / 128 threads
        int e = tid + i * 128;
        int r = e >> 3, kk = (e & 7) * 4;
        const float4 fv = *(const float4*)(A + (size_t)(m0 + r) * lda + kb + kk);
        v4h hv; hv.x = (_Float16)fv.x; hv.y = (_Float16)fv.y;
                hv.z = (_Float16)fv.z; hv.w = (_Float16)fv.w;
        *(v4h*)&As[r][kk] = hv;
      }
      if (kb + 32 < K && tid < 64)                // prefetch next K-slab
        __builtin_prefetch(A + (size_t)(m0 + tid) * lda + kb + 32, 0, 1);
    } else {
      for (int e = tid; e < 64 * 32; e += 128) {
        int r = e >> 5, kk = e & 31;
        int gr = m0 + r, gk = kb + kk;
        float v = (gr < M && gk < K) ? A[(size_t)gr * lda + gk] : 0.f;
        As[r][kk] = (_Float16)v;
      }
    }
    if (fullB) {
      #pragma unroll
      for (int i = 0; i < 4; ++i) {
        int e = tid + i * 128;
        int r = e >> 3, kk = (e & 7) * 4;
        const float4 fv = *(const float4*)(W + (size_t)(n0 + r) * K + kb + kk);
        v4h hv; hv.x = (_Float16)fv.x; hv.y = (_Float16)fv.y;
                hv.z = (_Float16)fv.z; hv.w = (_Float16)fv.w;
        *(v4h*)&Bs[r][kk] = hv;
      }
      if (kb + 32 < K && tid < 64)
        __builtin_prefetch(W + (size_t)(n0 + tid) * K + kb + 32, 0, 1);
    } else {
      for (int e = tid; e < 64 * 32; e += 128) {
        int r = e >> 5, kk = e & 31;
        int gn = n0 + r, gk = kb + kk;
        float v = (gn < N && gk < K) ? W[(size_t)gn * K + gk] : 0.f;
        Bs[r][kk] = (_Float16)v;
      }
    }
    __syncthreads();

    // A fragment: 16x32, lanes 0-15 K-halves {0-7,16-23}, lanes 16-31 {8-15,24-31}
    union { v16h v; unsigned u[8]; } af;
    const int mrow  = wave * 16 + (lane & 15);
    const int khalf = (lane >= 16) ? 8 : 0;
    #pragma unroll
    for (int p = 0; p < 8; ++p) {
      int k0 = ((p & 3) * 2) + ((p >= 4) ? 16 : 0) + khalf;
      af.u[p] = *(const unsigned*)&As[mrow][k0];
    }
    #pragma unroll
    for (int nt = 0; nt < 4; ++nt) {
      union { v16h v; unsigned u[8]; } bf;
      const int nrow = nt * 16 + (lane & 15);
      const int kb2  = (lane >= 16) ? 8 : 0;
      #pragma unroll
      for (int p = 0; p < 8; ++p)
        bf.u[p] = *(const unsigned*)&Bs[nrow][kb2 + ((p & 3) * 2) + ((p >= 4) ? 16 : 0)];
      acc[nt] = __builtin_amdgcn_wmma_f32_16x16x32_f16(
          false, af.v, false, bf.v, (short)0, acc[nt], false, false);
    }
    __syncthreads();
  }

  // C/D layout: lane 0-15 -> N=lane, M=r; lane 16-31 -> N=lane-16, M=8+r
  const int mbase = m0 + wave * 16 + ((lane >= 16) ? 8 : 0);
  const int ncol0 = n0 + (lane & 15);
  for (int nt = 0; nt < 4; ++nt) {
    int col = ncol0 + nt * 16;
    if (col >= N) continue;
    float bv = bias ? bias[col] : 0.f;
    for (int r = 0; r < 8; ++r) {
      int row = mbase + r;
      if (row >= M) continue;
      float v = acc[nt][r] + bv;
      if (epi == 1)      v = geluf(v);
      else if (epi == 2) v = fmaxf(v, 0.f);
      else if (epi == 3) v = sigmoidf(v);
      if (pos) v += pos[(size_t)(row % posT) * HID + col];
      C[(size_t)row * ldc + col] = v;
    }
  }
}

// ---------------------------------------------------------------------------
// Conv1d (k=3/5/7 split 42/42/44) + GELU. grid = chunkN*T, block = 128 (=channel)
// ---------------------------------------------------------------------------
__global__ void conv_kernel(const float* __restrict__ x,
                            const float* __restrict__ c3w, const float* __restrict__ c3b,
                            const float* __restrict__ c5w, const float* __restrict__ c5b,
                            const float* __restrict__ c7w, const float* __restrict__ c7b,
                            float* __restrict__ out, int T) {
  int nt = blockIdx.x;
  int n = nt / T, t = nt % T;
  int ch = threadIdx.x;
  const float* xb = x + (size_t)n * T * D_IN;
  const float* w; int ksz, pad; float acc;
  if (ch < 42)       { w = c3w + ch * 12;        ksz = 3; pad = 1; acc = c3b[ch]; }
  else if (ch < 84)  { w = c5w + (ch - 42) * 20; ksz = 5; pad = 2; acc = c5b[ch - 42]; }
  else               { w = c7w + (ch - 84) * 28; ksz = 7; pad = 3; acc = c7b[ch - 84]; }
  for (int k = 0; k < ksz; ++k) {
    int tp = t - pad + k;
    if (tp < 0 || tp >= T) continue;
    for (int d = 0; d < D_IN; ++d) acc += w[d * ksz + k] * xb[tp * D_IN + d];
  }
  out[(size_t)nt * HID + ch] = geluf(acc);
}

// ---------------------------------------------------------------------------
// GRU recurrence. One wave per block owns 16 sequences; W_hh pinned in LDS (f16),
// gh = h @ W_hh^T via v_wmma_f32_16x16x32_f16 each step (M16,K64,N192; 24 WMMA).
// gi = x@W_ih^T + b_ih precomputed in global (time-parallel GEMM).
// ---------------------------------------------------------------------------
__launch_bounds__(32)
__global__ void gru_kernel(const float* __restrict__ gi,     // (chunkN, T, 192)
                           const float* __restrict__ whh,    // (192, 64)
                           const float* __restrict__ bhh,    // (192)
                           float* __restrict__ feat,         // (chunkN, 128)
                           float* __restrict__ seq_out,      // (chunkN, T, 128) or null
                           int T, int reverse, int featOff) {
  __shared__ __align__(16) _Float16 wsh[192][68];
  __shared__ float    bsh[192];
  __shared__ float    h32[16][64];
  __shared__ __align__(16) _Float16 h16[16][68];
  __shared__ float    gh[16][192];
  const int lane = threadIdx.x;

  for (int e = lane; e < 192 * 16; e += 32) {     // 3072 float4s
    int r = e >> 4, k = (e & 15) * 4;
    const float4 fv = *(const float4*)(whh + (size_t)r * 64 + k);
    v4h hv; hv.x = (_Float16)fv.x; hv.y = (_Float16)fv.y;
            hv.z = (_Float16)fv.z; hv.w = (_Float16)fv.w;
    *(v4h*)&wsh[r][k] = hv;
  }
  for (int e = lane; e < 192; e += 32)      bsh[e] = bhh[e];
  for (int e = lane; e < 16 * 64; e += 32) { h32[e >> 6][e & 63] = 0.f; h16[e >> 6][e & 63] = (_Float16)0.f; }
  __syncthreads();

  const int seqbase = blockIdx.x * 16;
  for (int t = 0; t < T; ++t) {
    const int tt = reverse ? (T - 1 - t) : t;
    v8f acc[12];
    for (int i = 0; i < 12; ++i) for (int r = 0; r < 8; ++r) acc[i][r] = 0.f;
    #pragma unroll
    for (int kc = 0; kc < 2; ++kc) {
      union { v16h v; unsigned u[8]; } af;
      const int mrow  = lane & 15;
      const int khalf = (lane >= 16) ? 8 : 0;
      #pragma unroll
      for (int p = 0; p < 8; ++p) {
        int k0 = kc * 32 + ((p & 3) * 2) + ((p >= 4) ? 16 : 0) + khalf;
        af.u[p] = *(const unsigned*)&h16[mrow][k0];
      }
      #pragma unroll
      for (int nt = 0; nt < 12; ++nt) {
        union { v16h v; unsigned u[8]; } bf;
        const int nrow = nt * 16 + (lane & 15);
        const int kb2  = kc * 32 + ((lane >= 16) ? 8 : 0);
        #pragma unroll
        for (int p = 0; p < 8; ++p)
          bf.u[p] = *(const unsigned*)&wsh[nrow][kb2 + ((p & 3) * 2) + ((p >= 4) ? 16 : 0)];
        acc[nt] = __builtin_amdgcn_wmma_f32_16x16x32_f16(
            false, af.v, false, bf.v, (short)0, acc[nt], false, false);
      }
    }
    {
      const int mb = (lane >= 16) ? 8 : 0;
      const int nc = lane & 15;
      #pragma unroll
      for (int nt = 0; nt < 12; ++nt)
        for (int r = 0; r < 8; ++r) gh[mb + r][nt * 16 + nc] = acc[nt][r];
    }
    __syncthreads();
    for (int i = lane; i < 16 * 64; i += 32) {
      const int m = i >> 6, u = i & 63;
      const size_t gidx = ((size_t)(seqbase + m) * T + tt) * 192;
      float ir = gi[gidx + u], iz = gi[gidx + 64 + u], in = gi[gidx + 128 + u];
      float hr = gh[m][u] + bsh[u];
      float hz = gh[m][64 + u] + bsh[64 + u];
      float hn = gh[m][128 + u] + bsh[128 + u];
      float r  = sigmoidf(ir + hr);
      float z  = sigmoidf(iz + hz);
      float nn = tanhf(in + r * hn);
      float hv = (1.f - z) * nn + z * h32[m][u];
      h32[m][u] = hv;
      h16[m][u] = (_Float16)hv;
      if (seq_out)
        seq_out[((size_t)(seqbase + m) * T + tt) * HID + featOff + u] = hv;
    }
    __syncthreads();
  }
  for (int i = lane; i < 16 * 64; i += 32) {
    const int m = i >> 6, u = i & 63;
    feat[(size_t)(seqbase + m) * HID + featOff + u] = h32[m][u];
  }
}

// ---------------------------------------------------------------------------
// LayerNorm over D=128 (optional residual). In-place safe. grid=rows, block=128
// ---------------------------------------------------------------------------
__global__ void ln_kernel(const float* __restrict__ x, const float* __restrict__ res,
                          float* __restrict__ out, int D) {
  __shared__ float red[128];
  const int row = blockIdx.x, t = threadIdx.x;
  float v = x[(size_t)row * D + t] + (res ? res[(size_t)row * D + t] : 0.f);
  red[t] = v; __syncthreads();
  for (int s = 64; s > 0; s >>= 1) { if (t < s) red[t] += red[t + s]; __syncthreads(); }
  float mean = red[0] / (float)D; __syncthreads();
  float d = v - mean; red[t] = d * d; __syncthreads();
  for (int s = 64; s > 0; s >>= 1) { if (t < s) red[t] += red[t + s]; __syncthreads(); }
  float var = red[0] / (float)D;
  out[(size_t)row * D + t] = d * rsqrtf(var + LNEPS);
}

// ---------------------------------------------------------------------------
// Lane PointNet: relu(W2 @ relu(W1 @ pt + b1) + b2), max over 20 pts, * mask.
// grid = B*LL, block = 128
// ---------------------------------------------------------------------------
__global__ void lane_kernel(const float* __restrict__ lanes, const float* __restrict__ lmask,
                            const float* __restrict__ w1, const float* __restrict__ b1,
                            const float* __restrict__ w2, const float* __restrict__ b2,
                            float* __restrict__ lf) {
  __shared__ float h1[PP][HID];
  const int bl = blockIdx.x;
  const int o  = threadIdx.x;
  const float* pts = lanes + (size_t)bl * PP * 2;
  const float w1a = w1[o * 2], w1b = w1[o * 2 + 1], bb = b1[o];
  for (int p = 0; p < PP; ++p)
    h1[p][o] = fmaxf(0.f, w1a * pts[p * 2] + w1b * pts[p * 2 + 1] + bb);
  __syncthreads();
  float best = 0.f;
  for (int p = 0; p < PP; ++p) {
    float s = b2[o];
    for (int i = 0; i < HID; ++i) s += w2[o * HID + i] * h1[p][i];
    best = fmaxf(best, fmaxf(0.f, s));
  }
  lf[(size_t)bl * HID + o] = best * lmask[bl];
}

// ---------------------------------------------------------------------------
// Cross-attention core (2 heads, q-len 1, kv-len 50). grid=B, block=128
// ---------------------------------------------------------------------------
__global__ void ca_attn_kernel(const float* __restrict__ Q, const float* __restrict__ Kb,
                               const float* __restrict__ Vb, float* __restrict__ ctx) {
  __shared__ float qs[HID];
  __shared__ float sc[2][Tlen];
  const int b = blockIdx.x, t = threadIdx.x;
  qs[t] = Q[(size_t)b * HID + t];
  __syncthreads();
  if (t < 2 * Tlen) {
    int h = t / Tlen, k = t % Tlen;
    const float* kp = Kb + ((size_t)b * Tlen + k) * HID + h * 64;
    float s = 0.f;
    for (int d = 0; d < 64; ++d) s += qs[h * 64 + d] * kp[d];
    sc[h][k] = s * 0.125f;   // 1/sqrt(64)
  }
  __syncthreads();
  if (t < 2) {
    float mx = -1e30f;
    for (int k = 0; k < Tlen; ++k) mx = fmaxf(mx, sc[t][k]);
    float sum = 0.f;
    for (int k = 0; k < Tlen; ++k) { float e = expf(sc[t][k] - mx); sc[t][k] = e; sum += e; }
    float inv = 1.f / sum;
    for (int k = 0; k < Tlen; ++k) sc[t][k] *= inv;
  }
  __syncthreads();
  const int h = t >> 6;
  float s = 0.f;
  for (int k = 0; k < Tlen; ++k) s += sc[h][k] * Vb[((size_t)b * Tlen + k) * HID + t];
  ctx[(size_t)b * HID + t] = s;
}

__global__ void ca_pack_kernel(const float* __restrict__ veh, const float* __restrict__ ctxo,
                               float* __restrict__ cat) {
  const int b = blockIdx.x, t = threadIdx.x;
  cat[(size_t)b * 256 + t]       = veh[(size_t)b * HID + t];
  cat[(size_t)b * 256 + HID + t] = ctxo[(size_t)b * HID + t];
}

__global__ void ca_fuse_kernel(const float* __restrict__ veh, const float* __restrict__ gate,
                               const float* __restrict__ ctxo, const float* __restrict__ imask,
                               float* __restrict__ out) {
  const int b = blockIdx.x, t = threadIdx.x;
  size_t i = (size_t)b * HID + t;
  out[i] = veh[i] + gate[i] * ctxo[i] * imask[b];
}

// ---------------------------------------------------------------------------
// Node assembly (+type embedding) and validity vector. grid=B, block=128
// ---------------------------------------------------------------------------
__global__ void nodes_kernel(const float* __restrict__ fused, const float* __restrict__ nbr,
                             const float* __restrict__ infra, const float* __restrict__ lf,
                             const float* __restrict__ nmask, const float* __restrict__ imask,
                             const float* __restrict__ lmask, const float* __restrict__ temb,
                             float* __restrict__ nodes, float* __restrict__ valid) {
  const int b = blockIdx.x, ch = threadIdx.x;
  for (int j = 0; j < NNODE; ++j) {
    float v; int ty;
    if (j == 0)        { v = fused[(size_t)b * HID + ch]; ty = 0; }
    else if (j <= 32)  { v = nbr[((size_t)b * NB + (j - 1)) * HID + ch] * nmask[b * NB + j - 1]; ty = 1; }
    else if (j == 33)  { v = infra[(size_t)b * HID + ch] * imask[b]; ty = 2; }
    else               { v = lf[((size_t)b * LL + (j - 34)) * HID + ch]; ty = 3; }
    nodes[((size_t)b * NNODE + j) * HID + ch] = v + temb[ty * HID + ch];
  }
  if (ch < NNODE) {
    int j = ch; float v;
    if (j == 0)       v = 1.f;
    else if (j <= 32) v = nmask[b * NB + j - 1];
    else if (j == 33) v = imask[b];
    else              v = lmask[b * LL + (j - 34)];
    valid[(size_t)b * NNODE + j] = v;
  }
}

// ---------------------------------------------------------------------------
// GAT attention scores s1/s2. grid=B, block=256 (t = head*64 + node)
// ---------------------------------------------------------------------------
__global__ void gat_s_kernel(const float* __restrict__ Wh, const float* __restrict__ a1,
                             const float* __restrict__ a2, float* __restrict__ s1,
                             float* __restrict__ s2) {
  const int b = blockIdx.x, t = threadIdx.x;
  const int h = t >> 6, n = t & 63;
  const float* w = Wh + ((size_t)b * NNODE + n) * HID + h * 32;
  float x1 = 0.f, x2 = 0.f;
  for (int o = 0; o < 32; ++o) { x1 += w[o] * a1[h * 32 + o]; x2 += w[o] * a2[h * 32 + o]; }
  s1[((size_t)b * 4 + h) * NNODE + n] = x1;
  s2[((size_t)b * 4 + h) * NNODE + n] = x2;
}

// ---------------------------------------------------------------------------
// GAT leaky-relu attention + masked softmax + aggregate. grid=B*4, block=64
// ---------------------------------------------------------------------------
__global__ void gat_attn_kernel(const float* __restrict__ s1, const float* __restrict__ s2,
                                const float* __restrict__ valid, const float* __restrict__ Wh,
                                float* __restrict__ hcat) {
  __shared__ float alpha[NNODE][NNODE + 1];
  __shared__ float s1s[NNODE], s2s[NNODE], vs[NNODE];
  const int bh = blockIdx.x, b = bh >> 2, h = bh & 3;
  const int n = threadIdx.x;
  s1s[n] = s1[(size_t)bh * NNODE + n];
  s2s[n] = s2[(size_t)bh * NNODE + n];
  vs[n]  = valid[(size_t)b * NNODE + n];
  __syncthreads();
  if (vs[n] > 0.f) {
    float mx = -1e30f;
    for (int m = 0; m < NNODE; ++m)
      if (vs[m] > 0.f) { float e = s1s[n] + s2s[m]; e = e > 0.f ? e : 0.2f * e; mx = fmaxf(mx, e); }
    float sum = 0.f;
    for (int m = 0; m < NNODE; ++m) {
      float a = 0.f;
      if (vs[m] > 0.f) { float e = s1s[n] + s2s[m]; e = e > 0.f ? e : 0.2f * e; a = expf(e - mx); }
      alpha[n][m] = a; sum += a;
    }
    float inv = sum > 0.f ? 1.f / sum : 0.f;
    for (int m = 0; m < NNODE; ++m) alpha[n][m] *= inv;
  } else {
    for (int m = 0; m < NNODE; ++m) alpha[n][m] = 0.f;
  }
  __syncthreads();
  for (int o = 0; o < 32; ++o) {
    float s = 0.f;
    for (int m = 0; m < NNODE; ++m)
      s += alpha[n][m] * Wh[((size_t)b * NNODE + m) * HID + h * 32 + o];
    hcat[((size_t)b * NNODE + n) * HID + h * 32 + o] = s;
  }
}

// ---------------------------------------------------------------------------
// Final 6-way softmax. grid=B, block=32
// ---------------------------------------------------------------------------
__global__ void softmax6_kernel(const float* __restrict__ logits, float* __restrict__ out) {
  const int b = blockIdx.x;
  if (threadIdx.x == 0) {
    float mx = -1e30f;
    for (int i = 0; i < MODES; ++i) mx = fmaxf(mx, logits[b * MODES + i]);
    float e[MODES]; float s = 0.f;
    for (int i = 0; i < MODES; ++i) { e[i] = expf(logits[b * MODES + i] - mx); s += e[i]; }
    for (int i = 0; i < MODES; ++i) out[b * MODES + i] = e[i] / s;
  }
}

// ---------------------------------------------------------------------------
// Host orchestration
// ---------------------------------------------------------------------------
// Input index map (top-level setup_inputs order; params flattened as a JAX
// pytree: dict keys sorted, lists in order, each lin dict {b,w} sorted).
enum {
  IN_EGO = 0, IN_NBR, IN_INFRA, IN_LANES, IN_NMASK, IN_IMASK, IN_LMASK,
  CA_GATE_B = 7, CA_GATE_W, CA_K_B, CA_K_W, CA_OUT_B, CA_OUT_W, CA_Q_B, CA_Q_W, CA_V_B, CA_V_W,
  DEC1_B, DEC1_W, DEC2_B, DEC2_W,          // 17..20
  EGO_ENC0 = 21,                           // 21..37
  GNN0 = EGO_ENC0 + 17,                    // 38..44
  GNN1 = GNN0 + 7,                         // 45..51
  INPROJ_B = GNN1 + 7, INPROJ_W,           // 52,53
  INFRA_ENC0 = 54,                         // 54..70
  LANE1_B = INFRA_ENC0 + 17, LANE1_W, LANE2_B, LANE2_W,  // 71..74
  MODE_Bi, MODE_Wi,                        // 75,76
  NBR_ENC0 = 77,                           // 77..93
  PROB1_B = NBR_ENC0 + 17, PROB1_W, PROB2_B, PROB2_W, TYPE_EMBi  // 94..98
};
// encoder leaf offsets (sorted keys: c3b,c3w,c5b,c5w,c7b,c7w,gru_b{b_hh,b_ih,w_hh,w_ih},
// gru_f{...}, pos, proj{b,w})
enum { EC3B=0, EC3W, EC5B, EC5W, EC7B, EC7W,
       EGB_BHH, EGB_BIH, EGB_WHH, EGB_WIH,
       EGF_BHH, EGF_BIH, EGF_WHH, EGF_WIH,
       EPOS, EPROJ_B, EPROJ_W };
// gnn layer leaf offsets (W, a1, a2, ffn1{b,w}, ffn2{b,w})
enum { GW=0, GA1, GA2, GF1B, GF1W, GF2B, GF2W };

extern "C" void kernel_launch(void* const* d_in, const int* in_sizes, int n_in,
                              void* d_out, int out_size, void* d_ws, size_t ws_size,
                              hipStream_t stream) {
  (void)in_sizes; (void)n_in; (void)out_size; (void)ws_size;
  auto F = [&](int i) -> const float* { return (const float*)d_in[i]; };

  // ---- workspace layout (floats) ----
  float* w = (float*)d_ws;
  size_t off = 0;
  auto alloc = [&](size_t n) { float* p = w + off; off += n; return p; };

  float* ego_feat_raw   = alloc((size_t)Bsz * HID);
  float* ego_feat       = alloc((size_t)Bsz * HID);
  float* infra_feat_raw = alloc((size_t)Bsz * HID);
  float* infra_feat     = alloc((size_t)Bsz * HID);
  float* nbr_feat_raw   = alloc((size_t)Bsz * NB * HID);
  float* nbr_feat       = alloc((size_t)Bsz * NB * HID);
  float* infra_seq      = alloc((size_t)Bsz * Tlen * HID);
  float* lf             = alloc((size_t)Bsz * LL * HID);
  float* Qbuf           = alloc((size_t)Bsz * HID);
  float* ctxbuf         = alloc((size_t)Bsz * HID);
  float* ctxo           = alloc((size_t)Bsz * HID);
  float* catbuf         = alloc((size_t)Bsz * 256);
  float* gatebuf        = alloc((size_t)Bsz * HID);
  float* fusedbuf       = alloc((size_t)Bsz * HID);
  float* nodesbuf       = alloc((size_t)Bsz * NNODE * HID);
  float* validbuf       = alloc((size_t)Bsz * NNODE);
  float* hbuf           = alloc((size_t)Bsz * NNODE * HID);
  float* sf1            = alloc((size_t)Bsz * HID);
  float* sf2            = alloc((size_t)Bsz * HID);
  float* p1buf          = alloc((size_t)Bsz * 64);
  float* logitsbuf      = alloc((size_t)Bsz * MODES);

  // Aliased scratch zone (encoder chunk / CA K,V / GAT buffers; phases disjoint)
  float* scratch = alloc(5308416);
  float* c_buf  = scratch;                 // 128*50*128 = 819200
  float* c2_buf = scratch + 819200;
  float* gif    = scratch + 1638400;       // 128*50*192 = 1228800
  float* gib    = scratch + 2867200;
  float* Kbuf   = scratch;                 // CA phase
  float* Vbuf   = scratch + 819200;
  float* Whb    = scratch;                 // GAT phase
  float* hcat   = scratch + 1048576;
  float* ffnmid = scratch + 2097152;       // 8192*256
  float* ffnout = scratch + 4194304;
  float* s1b    = scratch + 5242880;
  float* s2b    = scratch + 5275648;

  const int T = Tlen;
#define GEMM(A, lda, Wp, bias, C, ldc, M, N, K, epi, pos, posT) \
  gemm_wmma_kernel<<<dim3(((N) + 63) / 64, ((M) + 63) / 64), 128, 0, stream>>>( \
      A, lda, Wp, bias, C, ldc, M, N, K, epi, pos, posT)

  // ---- temporal encoders (chunked, 128 sequences / chunk) ----
  auto run_encoder = [&](int EB, const float* x, int Nseq,
                         float* feat_raw, float* feat_ln, float* seq_out) {
    for (int s0 = 0; s0 < Nseq; s0 += 128) {
      const float* xc = x + (size_t)s0 * T * D_IN;
      conv_kernel<<<128 * T, 128, 0, stream>>>(
          xc, F(EB + EC3W), F(EB + EC3B), F(EB + EC5W), F(EB + EC5B),
          F(EB + EC7W), F(EB + EC7B), c_buf, T);
      // proj + gelu + positional (pos[row%T])
      GEMM(c_buf, HID, F(EB + EPROJ_W), F(EB + EPROJ_B), c2_buf, HID,
           128 * T, HID, HID, 1, F(EB + EPOS), T);
      // gi = x @ W_ih^T + b_ih (both directions, time-parallel)
      GEMM(c2_buf, HID, F(EB + EGF_WIH), F(EB + EGF_BIH), gif, 192,
           128 * T, 192, HID, 0, nullptr, 0);
      GEMM(c2_buf, HID, F(EB + EGB_WIH), F(EB + EGB_BIH), gib, 192,
           128 * T, 192, HID, 0, nullptr, 0);
      gru_kernel<<<8, 32, 0, stream>>>(
          gif, F(EB + EGF_WHH), F(EB + EGF_BHH), feat_raw + (size_t)s0 * HID,
          seq_out ? seq_out + (size_t)s0 * T * HID : nullptr, T, 0, 0);
      gru_kernel<<<8, 32, 0, stream>>>(
          gib, F(EB + EGB_WHH), F(EB + EGB_BHH), feat_raw + (size_t)s0 * HID,
          seq_out ? seq_out + (size_t)s0 * T * HID : nullptr, T, 1, GRU_H);
    }
    ln_kernel<<<Nseq, 128, 0, stream>>>(feat_raw, nullptr, feat_ln, HID);
  };

  run_encoder(EGO_ENC0,   F(IN_EGO),   Bsz,      ego_feat_raw,   ego_feat,   nullptr);
  run_encoder(NBR_ENC0,   F(IN_NBR),   Bsz * NB, nbr_feat_raw,   nbr_feat,   nullptr);
  run_encoder(INFRA_ENC0, F(IN_INFRA), Bsz,      infra_feat_raw, infra_feat, infra_seq);

  // ---- lane PointNet ----
  lane_kernel<<<Bsz * LL, 128, 0, stream>>>(
      F(IN_LANES), F(IN_LMASK), F(LANE1_W), F(LANE1_B), F(LANE2_W), F(LANE2_B), lf);

  // ---- cross attention (ego <- infra_seq) ----
  GEMM(ego_feat, HID, F(CA_Q_W), F(CA_Q_B), Qbuf, HID, Bsz, HID, HID, 0, nullptr, 0);
  GEMM(infra_seq, HID, F(CA_K_W), F(CA_K_B), Kbuf, HID, Bsz * T, HID, HID, 0, nullptr, 0);
  GEMM(infra_seq, HID, F(CA_V_W), F(CA_V_B), Vbuf, HID, Bsz * T, HID, HID, 0, nullptr, 0);
  ca_attn_kernel<<<Bsz, 128, 0, stream>>>(Qbuf, Kbuf, Vbuf, ctxbuf);
  GEMM(ctxbuf, HID, F(CA_OUT_W), F(CA_OUT_B), ctxo, HID, Bsz, HID, HID, 0, nullptr, 0);
  ca_pack_kernel<<<Bsz, 128, 0, stream>>>(ego_feat, ctxo, catbuf);
  GEMM(catbuf, 256, F(CA_GATE_W), F(CA_GATE_B), gatebuf, HID, Bsz, HID, 256, 3, nullptr, 0);
  ca_fuse_kernel<<<Bsz, 128, 0, stream>>>(ego_feat, gatebuf, ctxo, F(IN_IMASK), fusedbuf);
  ln_kernel<<<Bsz, 128, 0, stream>>>(fusedbuf, nullptr, fusedbuf, HID);

  // ---- node assembly + input projection ----
  nodes_kernel<<<Bsz, 128, 0, stream>>>(
      fusedbuf, nbr_feat, infra_feat, lf, F(IN_NMASK), F(IN_IMASK), F(IN_LMASK),
      F(TYPE_EMBi), nodesbuf, validbuf);
  GEMM(nodesbuf, HID, F(INPROJ_W), F(INPROJ_B), hbuf, HID,
       Bsz * NNODE, HID, HID, 0, nullptr, 0);

  // ---- GAT layers ----
  const int gnnBase[2] = { GNN0, GNN1 };
  for (int l = 0; l < 2; ++l) {
    const int G = gnnBase[l];
    GEMM(hbuf, HID, F(G + GW), nullptr, Whb, HID, Bsz * NNODE, HID, HID, 0, nullptr, 0);
    gat_s_kernel<<<Bsz, 256, 0, stream>>>(Whb, F(G + GA1), F(G + GA2), s1b, s2b);
    gat_attn_kernel<<<Bsz * 4, 64, 0, stream>>>(s1b, s2b, validbuf, Whb, hcat);
    ln_kernel<<<Bsz * NNODE, 128, 0, stream>>>(hcat, hbuf, hcat, HID);
    GEMM(hcat, HID, F(G + GF1W), F(G + GF1B), ffnmid, 256, Bsz * NNODE, 256, HID, 1, nullptr, 0);
    GEMM(ffnmid, 256, F(G + GF2W), F(G + GF2B), ffnout, HID, Bsz * NNODE, HID, 256, 0, nullptr, 0);
    ln_kernel<<<Bsz * NNODE, 128, 0, stream>>>(ffnout, hcat, hbuf, HID);
  }

  // ---- decoder (ctx = h[:,0,:] via strided A, lda = NNODE*HID) ----
  float* preds_out = (float*)d_out;                       // (B, 6*30*2) = (128, 360)
  float* probs_out = (float*)d_out + (size_t)Bsz * MODES * FUT * OUTD;
  GEMM(hbuf, NNODE * HID, F(DEC1_W), F(DEC1_B), sf1, HID, Bsz, HID, HID, 1, nullptr, 0);
  GEMM(sf1, HID, F(DEC2_W), F(DEC2_B), sf2, HID, Bsz, HID, HID, 1, nullptr, 0);
  GEMM(sf2, HID, F(MODE_Wi), F(MODE_Bi), preds_out, MODES * FUT * OUTD,
       Bsz, MODES * FUT * OUTD, HID, 0, nullptr, 0);
  GEMM(sf2, HID, F(PROB1_W), F(PROB1_B), p1buf, 64, Bsz, 64, HID, 1, nullptr, 0);
  GEMM(p1buf, 64, F(PROB2_W), F(PROB2_B), logitsbuf, MODES, Bsz, MODES, 64, 0, nullptr, 0);
  softmax6_kernel<<<Bsz, 32, 0, stream>>>(logitsbuf, probs_out);
#undef GEMM
}
